// HeteroEvolveGCN_10806137717433
// MI455X (gfx1250) — matrix operand order repaired and checked by reading, not compiled, not run on hardware
//
#include <hip/hip_runtime.h>

#define N_NODES 100000
#define D 128

typedef __attribute__((ext_vector_type(16))) __bf16 v16bf;
typedef __attribute__((ext_vector_type(8)))  float  v8f;

__device__ __forceinline__ __bf16 f2bf(float x) { return (__bf16)x; }
__device__ __forceinline__ float  bf2f(__bf16 x) { return (float)x; }

__device__ __forceinline__ v8f wmma_bf16(v16bf a, v16bf b, v8f c) {
  // v_wmma_f32_16x16x32_bf16  D = A*B + C   (probe-confirmed signature)
  return __builtin_amdgcn_wmma_f32_16x16x32_bf16(false, a, false, b, (short)0, c, false, false);
}

// A fragment (16x32 bf16, MxK) straight from a global f32 row.
// Layout: lane L: M = L%16, hi = L/16; a[j]=A[M][c*32+hi*8+j] j<8 ; a[8+j]=A[M][c*32+16+hi*8+j]
__device__ __forceinline__ v16bf load_afrag_f32(const float* __restrict__ Arow, int c, int hi) {
  const float* p0 = Arow + c * 32 + hi * 8;
  const float* p1 = p0 + 16;
  v16bf a;
#pragma unroll
  for (int j = 0; j < 8; ++j) { a[j] = f2bf(p0[j]); a[8 + j] = f2bf(p1[j]); }
  return a;
}

// B fragment (32x16 bf16, KxN) from LDS staged as B^T row-major [n][k].
// Layout: lane L: N = L%16, hi = L/16; b[j] = B[c*32 + hi*8 + j][N] j<8 ; b[8+j] = B[c*32 + 16 + hi*8 + j][N]
__device__ __forceinline__ v16bf load_bfrag(const __bf16* __restrict__ Bt, int n, int c, int hi) {
  const __bf16* p = Bt + n * D + c * 32 + hi * 8;
  v16bf b;
#pragma unroll
  for (int j = 0; j < 8; ++j) { b[j] = p[j]; b[8 + j] = p[16 + j]; }
  return b;
}

__device__ __forceinline__ float sigmoidf_(float x) { return 1.0f / (1.0f + __expf(-x)); }

// ---------------- elementwise / reduction kernels ----------------

__global__ void zero_kernel(float* __restrict__ p, int n) {
  int i = blockIdx.x * 256 + threadIdx.x;
  if (i < n) p[i] = 0.0f;
}

// wave-per-node LayerNorm over D=128 (lane handles 4 features)
__global__ __launch_bounds__(256) void ln_kernel(const float* __restrict__ x,
                                                 const float* __restrict__ g,
                                                 const float* __restrict__ b,
                                                 float* __restrict__ out, int n) {
  int wid = blockIdx.x * 8 + (threadIdx.x >> 5);
  int lane = threadIdx.x & 31;
  if (wid >= n) return;
  float4 v = ((const float4*)(x + (long)wid * D))[lane];
  float s = v.x + v.y + v.z + v.w;
#pragma unroll
  for (int o = 16; o > 0; o >>= 1) s += __shfl_xor(s, o, 32);
  float m = s * (1.0f / 128.0f);
  float d0 = v.x - m, d1 = v.y - m, d2 = v.z - m, d3 = v.w - m;
  float q = d0 * d0 + d1 * d1 + d2 * d2 + d3 * d3;
#pragma unroll
  for (int o = 16; o > 0; o >>= 1) q += __shfl_xor(q, o, 32);
  float inv = rsqrtf(q * (1.0f / 128.0f) + 1e-5f);
  float4 gv = ((const float4*)g)[lane];
  float4 bv = ((const float4*)b)[lane];
  float4 o4;
  o4.x = d0 * inv * gv.x + bv.x;
  o4.y = d1 * inv * gv.y + bv.y;
  o4.z = d2 * inv * gv.z + bv.z;
  o4.w = d3 * inv * gv.w + bv.w;
  ((float4*)(out + (long)wid * D))[lane] = o4;
}

__global__ void deg_kernel(const int* __restrict__ dst, float* __restrict__ deg, int e) {
  int i = blockIdx.x * 256 + threadIdx.x;
  if (i < e) atomicAdd(&deg[dst[i]], 1.0f);
}

__global__ void dinv_kernel(float* __restrict__ deg, int n) {
  int i = blockIdx.x * 256 + threadIdx.x;
  if (i < n) deg[i] = rsqrtf(deg[i] + 1.0f);  // +1 self-loop; always >= 1
}

// scores[i] = (h[i] . p) * rsqrt(p.p + 1e-12), wave per node
__global__ __launch_bounds__(256) void score_kernel(const float* __restrict__ h,
                                                    const float* __restrict__ p,
                                                    float* __restrict__ scores, int n) {
  int wid = blockIdx.x * 8 + (threadIdx.x >> 5);
  int lane = threadIdx.x & 31;
  if (wid >= n) return;
  float4 pv = ((const float4*)p)[lane];
  float4 hv = ((const float4*)(h + (long)wid * D))[lane];
  float dt = hv.x * pv.x + hv.y * pv.y + hv.z * pv.z + hv.w * pv.w;
  float pp = pv.x * pv.x + pv.y * pv.y + pv.z * pv.z + pv.w * pv.w;
#pragma unroll
  for (int o = 16; o > 0; o >>= 1) { dt += __shfl_xor(dt, o, 32); pp += __shfl_xor(pp, o, 32); }
  scores[wid] = dt * rsqrtf(pp + 1e-12f);
}

// single-block iterative top-128 argmax (ties -> lowest index, matches lax.top_k)
__global__ __launch_bounds__(1024) void topk_kernel(float* __restrict__ scores, int n,
                                                    int* __restrict__ idx_out,
                                                    float* __restrict__ val_out) {
  __shared__ float sval[32];
  __shared__ int sidx[32];
  int tid = threadIdx.x, lane = tid & 31, w = tid >> 5;
  for (int iter = 0; iter < 128; ++iter) {
    float best = -3.4e38f;
    int bidx = 0x7fffffff;
    for (int i = tid; i < n; i += 1024) {
      float v = scores[i];
      if (v > best || (v == best && i < bidx)) { best = v; bidx = i; }
    }
#pragma unroll
    for (int o = 16; o > 0; o >>= 1) {
      float ov = __shfl_xor(best, o, 32);
      int oi = __shfl_xor(bidx, o, 32);
      if (ov > best || (ov == best && oi < bidx)) { best = ov; bidx = oi; }
    }
    if (lane == 0) { sval[w] = best; sidx[w] = bidx; }
    __syncthreads();
    if (w == 0) {
      float v2 = sval[lane];
      int i2 = sidx[lane];
#pragma unroll
      for (int o = 16; o > 0; o >>= 1) {
        float ov = __shfl_xor(v2, o, 32);
        int oi = __shfl_xor(i2, o, 32);
        if (ov > v2 || (ov == v2 && oi < i2)) { v2 = ov; i2 = oi; }
      }
      if (lane == 0) { idx_out[iter] = i2; val_out[iter] = v2; scores[i2] = -3.4e38f; }
    }
    __syncthreads();
  }
}

// agg init with self-loop term: agg[i] = h[i] * dinv[i]^2
__global__ void selfloop_kernel(const float* __restrict__ h, const float* __restrict__ dinv,
                                float* __restrict__ agg, int n) {
  long i = (long)blockIdx.x * 256 + threadIdx.x;
  if (i >= (long)n * D) return;
  float c = dinv[i >> 7];
  agg[i] = h[i] * c * c;
}

// wave per edge: agg[dst] += h[src] * dinv[src]*dinv[dst]   (L2-resident atomics)
__global__ __launch_bounds__(256) void edge_agg_kernel(const float* __restrict__ h,
                                                       const int* __restrict__ src,
                                                       const int* __restrict__ dst,
                                                       const float* __restrict__ dinv,
                                                       float* __restrict__ agg, int e) {
  int wid = blockIdx.x * 8 + (threadIdx.x >> 5);
  int lane = threadIdx.x & 31;
  if (wid >= e) return;
  int s = src[wid], d = dst[wid];
  float coef = dinv[s] * dinv[d];
  float4 v = ((const float4*)(h + (long)s * D))[lane];
  float* ad = agg + (long)d * D + lane * 4;
  atomicAdd(ad + 0, v.x * coef);
  atomicAdd(ad + 1, v.y * coef);
  atomicAdd(ad + 2, v.z * coef);
  atomicAdd(ad + 3, v.w * coef);
}

// ---------------- WMMA GEMM: C[nrows x NCOLS] = A[nrows x 128] @ B[128 x NCOLS] ----------------
// 8 waves/block, wave = 16-row strip, all NCOLS/16 tiles; B^T staged bf16 in LDS.
template <int NCOLS, bool RELU, bool BIAS, bool TRANSB>
__global__ __launch_bounds__(256) void matmul_kernel(const float* __restrict__ A,
                                                     const float* __restrict__ Bg,
                                                     const float* __restrict__ bias,
                                                     float* __restrict__ C, int nrows) {
  __shared__ __bf16 Bt[NCOLS * D];
  const int tid = threadIdx.x;
  for (int i = tid; i < NCOLS * D; i += 256) {
    int n = i >> 7, k = i & 127;
    float v = TRANSB ? Bg[n * D + k] : Bg[k * NCOLS + n];
    Bt[i] = f2bf(v);
  }
  __syncthreads();

  const int w = tid >> 5, lane = tid & 31;
  const int hi = lane >> 4, lcol = lane & 15;
  int row = blockIdx.x * 128 + w * 16 + lcol;
  int arow = row < nrows ? row : (nrows - 1);  // clamp; stores are guarded
  const float* Arow = A + (long)arow * D;

  constexpr int NT = NCOLS / 16;
  v8f acc[NT] = {};
#pragma unroll
  for (int c = 0; c < 4; ++c) {
    v16bf a = load_afrag_f32(Arow, c, hi);
#pragma unroll
    for (int t = 0; t < NT; ++t) {
      v16bf b = load_bfrag(Bt, t * 16 + lcol, c, hi);
      acc[t] = wmma_bf16(a, b, acc[t]);
    }
  }
#pragma unroll
  for (int t = 0; t < NT; ++t) {
#pragma unroll
    for (int r = 0; r < 8; ++r) {
      int orow = blockIdx.x * 128 + w * 16 + r + 8 * hi;
      int ocol = t * 16 + lcol;
      if (orow < nrows) {
        float v = acc[t][r];
        if (BIAS) v += bias[ocol];
        if (RELU) v = v > 0.0f ? v : 0.0f;
        C[(long)orow * NCOLS + ocol] = v;
      }
    }
  }
}

// ---------------- fused EvolveGCN-H matrix GRU (one 256-thread block) ----------------
// Z[f][k] = h[idx[k]][f]*tanh(vals[k]);  z=sig(Wz@Z+Uz@W+bz); r=sig(Wr@Z+Ur@W+br)
// hc=tanh(Wh@Z+Uh@(r*W)+bh);  Wout=(1-z)*W+z*hc.   All GEMMs 128x128x128 via WMMA.
__global__ __launch_bounds__(256) void evolve_kernel(
    const float* __restrict__ h, const float* __restrict__ W,
    const int* __restrict__ idx, const float* __restrict__ vals,
    const float* __restrict__ Wz, const float* __restrict__ Uz, const float* __restrict__ bz,
    const float* __restrict__ Wr, const float* __restrict__ Ur, const float* __restrict__ br,
    const float* __restrict__ Wh, const float* __restrict__ Uh, const float* __restrict__ bh,
    float* __restrict__ Wout) {
  __shared__ __bf16 Bt[D * D];  // 32 KB: holds Z^T, then W^T, then (r*W)^T
  const int tid = threadIdx.x;
  const int w = tid >> 5, lane = tid & 31;
  const int hi = lane >> 4, lcol = lane & 15;
  const int rowoff = (w * 16 + lcol) * D;  // A row for this lane (M = lane%16)

  // Bt = Z^T: row k = h[idx[k]] * tanh(vals[k])
  for (int i = tid; i < D * D; i += 256) {
    int k = i >> 7, f = i & 127;
    Bt[i] = f2bf(h[(long)idx[k] * D + f] * tanhf(vals[k]));
  }
  __syncthreads();

  v8f accZ[8] = {}, accR[8] = {}, accH[8] = {};
#pragma unroll
  for (int c = 0; c < 4; ++c) {
    v16bf az = load_afrag_f32(Wz + rowoff, c, hi);
    v16bf ar = load_afrag_f32(Wr + rowoff, c, hi);
    v16bf ah = load_afrag_f32(Wh + rowoff, c, hi);
#pragma unroll
    for (int t = 0; t < 8; ++t) {
      v16bf b = load_bfrag(Bt, t * 16 + lcol, c, hi);
      accZ[t] = wmma_bf16(az, b, accZ[t]);
      accR[t] = wmma_bf16(ar, b, accR[t]);
      accH[t] = wmma_bf16(ah, b, accH[t]);
    }
  }
  __syncthreads();
  // Bt = W^T
  for (int i = tid; i < D * D; i += 256) {
    int n = i >> 7, k = i & 127;
    Bt[i] = f2bf(W[k * D + n]);
  }
  __syncthreads();
#pragma unroll
  for (int c = 0; c < 4; ++c) {
    v16bf az = load_afrag_f32(Uz + rowoff, c, hi);
    v16bf ar = load_afrag_f32(Ur + rowoff, c, hi);
#pragma unroll
    for (int t = 0; t < 8; ++t) {
      v16bf b = load_bfrag(Bt, t * 16 + lcol, c, hi);
      accZ[t] = wmma_bf16(az, b, accZ[t]);
      accR[t] = wmma_bf16(ar, b, accR[t]);
    }
  }
  // gates (kept in registers)
#pragma unroll
  for (int t = 0; t < 8; ++t)
#pragma unroll
    for (int r = 0; r < 8; ++r) {
      int irow = w * 16 + r + 8 * hi;
      accZ[t][r] = sigmoidf_(accZ[t][r] + bz[irow]);
      accR[t][r] = sigmoidf_(accR[t][r] + br[irow]);
    }
  __syncthreads();  // all W^T reads complete before in-place scale
  // Bt := (r * W)^T   (each wave updates its own disjoint tile region)
#pragma unroll
  for (int t = 0; t < 8; ++t)
#pragma unroll
    for (int r = 0; r < 8; ++r) {
      int irow = w * 16 + r + 8 * hi;
      int jcol = t * 16 + lcol;
      int o = jcol * D + irow;
      Bt[o] = f2bf(bf2f(Bt[o]) * accR[t][r]);
    }
  __syncthreads();
#pragma unroll
  for (int c = 0; c < 4; ++c) {
    v16bf ah = load_afrag_f32(Uh + rowoff, c, hi);
#pragma unroll
    for (int t = 0; t < 8; ++t) {
      v16bf b = load_bfrag(Bt, t * 16 + lcol, c, hi);
      accH[t] = wmma_bf16(ah, b, accH[t]);
    }
  }
  // Wout = (1-z)*W + z*tanh(accH + bh)
#pragma unroll
  for (int t = 0; t < 8; ++t)
#pragma unroll
    for (int r = 0; r < 8; ++r) {
      int irow = w * 16 + r + 8 * hi;
      int jcol = t * 16 + lcol;
      float hc = tanhf(accH[t][r] + bh[irow]);
      float z = accZ[t][r];
      Wout[irow * D + jcol] = (1.0f - z) * W[irow * D + jcol] + z * hc;
    }
}

// ---------------- launch ----------------

extern "C" void kernel_launch(void* const* d_in, const int* in_sizes, int n_in,
                              void* d_out, int out_size, void* d_ws, size_t ws_size,
                              hipStream_t stream) {
  (void)n_in; (void)out_size; (void)ws_size;
  const float* x     = (const float*)d_in[0];
  const int*   eidx  = (const int*)d_in[1];
  const float* ln1_g = (const float*)d_in[2];
  const float* ln1_b = (const float*)d_in[3];
  const float* p0    = (const float*)d_in[4];
  const float* W0    = (const float*)d_in[5];
  const float* Wz0   = (const float*)d_in[6];
  const float* Uz0   = (const float*)d_in[7];
  const float* Wr0   = (const float*)d_in[8];
  const float* Ur0   = (const float*)d_in[9];
  const float* Wh0   = (const float*)d_in[10];
  const float* Uh0   = (const float*)d_in[11];
  const float* bz0   = (const float*)d_in[12];
  const float* br0   = (const float*)d_in[13];
  const float* bh0   = (const float*)d_in[14];
  const float* p1    = (const float*)d_in[15];
  const float* W1    = (const float*)d_in[16];
  const float* Wz1   = (const float*)d_in[17];
  const float* Uz1   = (const float*)d_in[18];
  const float* Wr1   = (const float*)d_in[19];
  const float* Ur1   = (const float*)d_in[20];
  const float* Wh1   = (const float*)d_in[21];
  const float* Uh1   = (const float*)d_in[22];
  const float* bz1   = (const float*)d_in[23];
  const float* br1   = (const float*)d_in[24];
  const float* bh1   = (const float*)d_in[25];
  const float* ln2_g = (const float*)d_in[26];
  const float* ln2_b = (const float*)d_in[27];
  const float* fcW   = (const float*)d_in[28];
  const float* fcb   = (const float*)d_in[29];

  const int N = in_sizes[0] / D;       // 100000
  const int E = in_sizes[1] / 2;       // 1600000
  const int* src = eidx;
  const int* dst = eidx + E;

  float* ws    = (float*)d_ws;
  float* bufA  = ws;                        // N*128
  float* bufB  = bufA + (size_t)N * D;      // N*128
  float* deg   = bufB + (size_t)N * D;      // N  (becomes dinv)
  float* score = deg + N;                   // N
  float* tvals = score + N;                 // 128
  int*   tidx  = (int*)(tvals + 128);       // 128
  float* W0t   = (float*)(tidx + 128);      // 128*128
  float* W1t   = W0t + D * D;               // 128*128

  const int wv_blocks   = (N + 7) / 8;
  const int e_wv_blocks = (E + 7) / 8;
  const int mm_blocks   = (N + 127) / 128;

  // degrees (shared by both layers)
  zero_kernel<<<(N + 255) / 256, 256, 0, stream>>>(deg, N);
  deg_kernel<<<(E + 255) / 256, 256, 0, stream>>>(dst, deg, E);
  dinv_kernel<<<(N + 255) / 256, 256, 0, stream>>>(deg, N);

  // layer 1
  ln_kernel<<<wv_blocks, 256, 0, stream>>>(x, ln1_g, ln1_b, bufA, N);
  score_kernel<<<wv_blocks, 256, 0, stream>>>(bufA, p0, score, N);
  topk_kernel<<<1, 1024, 0, stream>>>(score, N, tidx, tvals);
  evolve_kernel<<<1, 256, 0, stream>>>(bufA, W0, tidx, tvals, Wz0, Uz0, bz0, Wr0, Ur0, br0,
                                       Wh0, Uh0, bh0, W0t);
  selfloop_kernel<<<(int)(((long)N * D + 255) / 256), 256, 0, stream>>>(bufA, deg, bufB, N);
  edge_agg_kernel<<<e_wv_blocks, 256, 0, stream>>>(bufA, src, dst, deg, bufB, E);
  matmul_kernel<128, true, false, false><<<mm_blocks, 256, 0, stream>>>(bufB, W0t, nullptr, bufA, N);

  // layer 2
  score_kernel<<<wv_blocks, 256, 0, stream>>>(bufA, p1, score, N);
  topk_kernel<<<1, 1024, 0, stream>>>(score, N, tidx, tvals);
  evolve_kernel<<<1, 256, 0, stream>>>(bufA, W1, tidx, tvals, Wz1, Uz1, bz1, Wr1, Ur1, br1,
                                       Wh1, Uh1, bh1, W1t);
  selfloop_kernel<<<(int)(((long)N * D + 255) / 256), 256, 0, stream>>>(bufA, deg, bufB, N);
  edge_agg_kernel<<<e_wv_blocks, 256, 0, stream>>>(bufA, src, dst, deg, bufB, E);
  matmul_kernel<128, false, false, false><<<mm_blocks, 256, 0, stream>>>(bufB, W1t, nullptr, bufA, N);

  // LN2 + FC
  ln_kernel<<<wv_blocks, 256, 0, stream>>>(bufA, ln2_g, ln2_b, bufB, N);
  matmul_kernel<64, false, true, true><<<mm_blocks, 256, 0, stream>>>(bufB, fcW, fcb,
                                                                      (float*)d_out, N);
}
